// MultiModalTeacherModel_16716012716319
// MI455X (gfx1250) — matrix operand order repaired and checked by reading, not compiled, not run on hardware
//
#include <hip/hip_runtime.h>
#include <stdint.h>

// ---------------------------------------------------------------------------
// MI455X (gfx1250) implementation of the multimodal transformer teacher.
// All GEMMs in bf16 WMMA (v_wmma_f32_16x16x32_bf16, fp32 accum).
// Tile staging via async global->LDS (ASYNCcnt) where beneficial.
// ---------------------------------------------------------------------------

typedef __bf16 bf16;
typedef __bf16 v16bf __attribute__((ext_vector_type(16)));
typedef float  v8f   __attribute__((ext_vector_type(8)));

#ifndef USE_ASYNC_LDS
#define USE_ASYNC_LDS 1
#endif

// 16-byte global -> LDS copy. On gfx1250 use the async-to-LDS path (tracked
// by ASYNCcnt); LDS aperture low 32 bits are the LDS byte offset.
__device__ inline void cp16(void* lds_dst, const void* gsrc) {
#if USE_ASYNC_LDS
  uint32_t loff = (uint32_t)(uintptr_t)lds_dst;
  uint64_t gaddr = (uint64_t)(uintptr_t)gsrc;
  asm volatile("global_load_async_to_lds_b128 %0, %1, off"
               :: "v"(loff), "v"(gaddr) : "memory");
#else
  *(uint4*)lds_dst = *(const uint4*)gsrc;
#endif
}
__device__ inline void async_wait0() {
#if USE_ASYNC_LDS
  asm volatile("s_wait_asynccnt 0" ::: "memory");
#endif
}

__device__ inline float gelu_f(float x) {
  return 0.5f * x * (1.f + erff(x * 0.70710678118654752f));
}
__device__ inline float wave_sum(float v) {
  #pragma unroll
  for (int o = 16; o; o >>= 1) v += __shfl_xor(v, o, 32);
  return v;
}
__device__ inline float wave_max(float v) {
  #pragma unroll
  for (int o = 16; o; o >>= 1) v = fmaxf(v, __shfl_xor(v, o, 32));
  return v;
}

// ---- WMMA fragment helpers (contiguous k order, k = 16*(lane/16) + element, same in A and B) ------------------
// A fragment: 16x32 bf16 tile, row-major with leading dim ld.
__device__ inline v16bf load_a_frag(const bf16* A, int ld) {
  const int lane = threadIdx.x & 31;
  const int m  = lane & 15;
  const int k0 = (lane & 16) ? 16 : 0;  // VGPR0-3: K 0..7 (lo lanes) / 16..23
  const int k1 = k0 + 8;                // VGPR4-7: K 8..15 / 24..31
  v16bf a;
  #pragma unroll
  for (int i = 0; i < 8; ++i) a[i]     = A[m * ld + k0 + i];
  #pragma unroll
  for (int i = 0; i < 8; ++i) a[8 + i] = A[m * ld + k1 + i];
  return a;
}
// B fragment built from W stored (N,K) row-major: B[k][n] = W[n][k].
__device__ inline v16bf load_b_nk(const bf16* W, int ld) {
  const int lane = threadIdx.x & 31;
  const int n  = lane & 15;
  const int kb = (lane & 16) ? 16 : 0;
  v16bf b;
  #pragma unroll
  for (int i = 0; i < 16; ++i) b[i] = W[n * ld + kb + i];
  return b;
}
// B fragment from a (K,N) row-major tile: B[k][n] = S[k*ld + n].
__device__ inline v16bf load_b_kn(const bf16* S, int ld) {
  const int lane = threadIdx.x & 31;
  const int n  = lane & 15;
  const int kb = (lane & 16) ? 16 : 0;
  v16bf b;
  #pragma unroll
  for (int i = 0; i < 16; ++i) b[i] = S[(kb + i) * ld + n];
  return b;
}
__device__ inline v8f wmma_bf16(v16bf a, v16bf b, v8f c) {
  return __builtin_amdgcn_wmma_f32_16x16x32_bf16(false, a, false, b,
                                                 (short)0, c, false, false);
}

// ---------------------------------------------------------------------------
// Param conversion f32 -> bf16
// ---------------------------------------------------------------------------
__global__ void cvt_bf16_kernel(const float* __restrict__ s, bf16* __restrict__ d, int n) {
  int i = blockIdx.x * 256 + threadIdx.x;
  if (i < n) d[i] = (bf16)s[i];
}
// center tap of conv (E, E, 3) -> (E, E) bf16
__global__ void cvt_center_kernel(const float* __restrict__ s, bf16* __restrict__ d) {
  int i = blockIdx.x * 256 + threadIdx.x;  // 65536
  d[i] = (bf16)s[i * 3 + 1];
}

// ---------------------------------------------------------------------------
// Kernel 1: fused skeleton embed + joint attention + mean over joints.
// One block per (b,s); writes one row of comb (B, S+T, E).
// ---------------------------------------------------------------------------
__global__ void skel_fused_kernel(
    const float* __restrict__ skel,                        // (4096, 32, 3)
    const float* __restrict__ c1w, const float* __restrict__ c1b,
    const float* __restrict__ bn1g, const float* __restrict__ bn1b,
    const float* __restrict__ bn1m, const float* __restrict__ bn1v,
    const bf16*  __restrict__ w2c, const float* __restrict__ c2b,
    const float* __restrict__ bn2g, const float* __restrict__ bn2b,
    const float* __restrict__ bn2m, const float* __restrict__ bn2v,
    const bf16*  __restrict__ jawi, const float* __restrict__ jabi,
    const bf16*  __restrict__ jawo, const float* __restrict__ jabo,
    bf16* __restrict__ comb)                               // (32, 256, 256)
{
  extern __shared__ char smem[];
  bf16*  h0  = (bf16*)smem;                    // 32x256
  bf16*  hb  = h0 + 32 * 256;                  // 32x256
  bf16*  qb  = hb + 32 * 256;                  // 32x256 (scaled)
  bf16*  kb  = qb + 32 * 256;                  // 32x256
  bf16*  vb  = kb + 32 * 256;                  // 32x256
  bf16*  ob  = vb + 32 * 256;                  // 32x256
  bf16*  pb  = ob + 32 * 256;                  // 4 heads x 32x32 probs
  float* sc  = (float*)(pb + 4 * 32 * 32);     // 4 heads x 32x32 scores
  float* msum = sc + 4 * 32 * 32;              // 256 column accumulators
  float* xin = msum + 256;                     // 32x3 input

  const int bs = blockIdx.x;
  const int tid = threadIdx.x, wave = tid >> 5, lane = tid & 31;
  if (tid < 96) xin[tid] = skel[(size_t)bs * 96 + tid];
  msum[tid] = 0.f;
  __syncthreads();

  // --- embed layer 1 (conv center tap, K=3) + BN1 + GELU -> h0 (bf16) ---
  for (int idx = tid; idx < 32 * 256; idx += 256) {
    int j = idx >> 8, e = idx & 255;
    float s = c1b[e];
    #pragma unroll
    for (int c = 0; c < 3; ++c) s += xin[j * 3 + c] * c1w[e * 9 + c * 3 + 1];
    s = (s - bn1m[e]) * rsqrtf(bn1v[e] + 1e-5f) * bn1g[e] + bn1b[e];
    h0[idx] = (bf16)gelu_f(s);
  }
  __syncthreads();

  // --- embed layer 2: h = gelu(bn2(h0 @ W2^T + b2)), 32x256x256 WMMA ---
  for (int it = 0; it < 4; ++it) {
    int t = wave * 4 + it, mt = t & 1, nt = t >> 1;
    v8f acc = {};
    for (int kk = 0; kk < 8; ++kk) {
      v16bf a = load_a_frag(h0 + mt * 16 * 256 + kk * 32, 256);
      v16bf b = load_b_nk(w2c + nt * 16 * 256 + kk * 32, 256);
      acc = wmma_bf16(a, b, acc);
    }
    int n = lane & 15, mb = (lane & 16) ? 8 : 0, gn = nt * 16 + n;
    float rs = rsqrtf(bn2v[gn] + 1e-5f);
    #pragma unroll
    for (int r = 0; r < 8; ++r) {
      float v = acc[r] + c2b[gn];
      v = (v - bn2m[gn]) * rs * bn2g[gn] + bn2b[gn];
      hb[(mt * 16 + mb + r) * 256 + gn] = (bf16)gelu_f(v);
    }
  }
  __syncthreads();

  // --- joint attention QKV: 32x768x256 WMMA (scale folded into q) ---
  for (int it = 0; it < 12; ++it) {
    int t = wave * 12 + it, mt = t & 1, nt = t >> 1;
    v8f acc = {};
    for (int kk = 0; kk < 8; ++kk) {
      v16bf a = load_a_frag(hb + mt * 16 * 256 + kk * 32, 256);
      v16bf b = load_b_nk(jawi + nt * 16 * 256 + kk * 32, 256);
      acc = wmma_bf16(a, b, acc);
    }
    int n = lane & 15, mb = (lane & 16) ? 8 : 0, col = nt * 16 + n;
    #pragma unroll
    for (int r = 0; r < 8; ++r) {
      int m = mt * 16 + mb + r;
      float v = acc[r] + jabi[col];
      if (col < 256)      qb[m * 256 + col] = (bf16)(v * 0.125f);  // 1/sqrt(64)
      else if (col < 512) kb[m * 256 + (col - 256)] = (bf16)v;
      else                vb[m * 256 + (col - 512)] = (bf16)v;
    }
  }
  __syncthreads();

  // --- scores per head (4 heads, d=64): S_h = q_h @ k_h^T ---
  for (int it = 0; it < 2; ++it) {
    int t = wave * 2 + it, h = t >> 2, tt = t & 3, mt = tt >> 1, nt = tt & 1;
    v8f acc = {};
    for (int kk = 0; kk < 2; ++kk) {
      v16bf a = load_a_frag(qb + mt * 16 * 256 + h * 64 + kk * 32, 256);
      v16bf b = load_b_nk(kb + nt * 16 * 256 + h * 64 + kk * 32, 256);
      acc = wmma_bf16(a, b, acc);
    }
    int n = lane & 15, mb = (lane & 16) ? 8 : 0;
    #pragma unroll
    for (int r = 0; r < 8; ++r)
      sc[h * 1024 + (mt * 16 + mb + r) * 32 + nt * 16 + n] = acc[r];
  }
  __syncthreads();

  // --- softmax over 32 keys, 128 rows (4 heads x 32) ---
  if (tid < 128) {
    int h = tid >> 5, r = tid & 31;
    float* row = sc + h * 1024 + r * 32;
    float mx = -1e30f;
    for (int c = 0; c < 32; ++c) mx = fmaxf(mx, row[c]);
    float e[32], sum = 0.f;
    for (int c = 0; c < 32; ++c) { e[c] = __expf(row[c] - mx); sum += e[c]; }
    float inv = 1.f / sum;
    for (int c = 0; c < 32; ++c) pb[h * 1024 + r * 32 + c] = (bf16)(e[c] * inv);
  }
  __syncthreads();

  // --- O_h = P_h @ V_h (K=32, one WMMA per tile) ---
  for (int it = 0; it < 4; ++it) {
    int t = wave * 4 + it, h = t >> 3, tt = t & 7, mt = tt >> 2, nt = tt & 3;
    v16bf a = load_a_frag(pb + h * 1024 + mt * 16 * 32, 32);
    v16bf b = load_b_kn(vb + h * 64 + nt * 16, 256);
    v8f acc = {};
    acc = wmma_bf16(a, b, acc);
    int n = lane & 15, mb = (lane & 16) ? 8 : 0;
    #pragma unroll
    for (int r = 0; r < 8; ++r)
      ob[(mt * 16 + mb + r) * 256 + h * 64 + nt * 16 + n] = (bf16)acc[r];
  }
  __syncthreads();

  // --- out proj fused with mean over joints: ds_add_f32 column reduce ---
  for (int it = 0; it < 4; ++it) {
    int t = wave * 4 + it, mt = t & 1, nt = t >> 1;
    v8f acc = {};
    for (int kk = 0; kk < 8; ++kk) {
      v16bf a = load_a_frag(ob + mt * 16 * 256 + kk * 32, 256);
      v16bf b = load_b_nk(jawo + nt * 16 * 256 + kk * 32, 256);
      acc = wmma_bf16(a, b, acc);
    }
    float part = 0.f;
    #pragma unroll
    for (int r = 0; r < 8; ++r) part += acc[r];
    int n = lane & 15;
    atomicAdd(&msum[nt * 16 + n], part);
  }
  __syncthreads();
  {
    int b = bs >> 7, s = bs & 127;
    comb[((size_t)(b * 256 + s)) * 256 + tid] =
        (bf16)(msum[tid] * (1.f / 32.f) + jabo[tid]);
  }
}

// ---------------------------------------------------------------------------
// Kernel 2a: accelerometer branch (Linear(4->256)+LN+GELU, Linear+LN+GELU).
// One block per 32 rows.
// ---------------------------------------------------------------------------
__global__ void acc_branch_kernel(
    const float* __restrict__ ain,                            // (4096, 4)
    const float* __restrict__ w1, const float* __restrict__ b1,
    const float* __restrict__ l1g, const float* __restrict__ l1b,
    const bf16*  __restrict__ w2, const float* __restrict__ b2,
    const float* __restrict__ l2g, const float* __restrict__ l2b,
    bf16* __restrict__ zout)                                  // (4096, 256)
{
  extern __shared__ char smem[];
  float* u  = (float*)smem;                  // 32x256 f32
  bf16*  z1 = (bf16*)(smem + 32 * 256 * 4);  // 32x256 bf16
  float* ax = (float*)(smem + 32 * 256 * 4 + 32 * 256 * 2);  // 128 f32
  const int r0 = blockIdx.x * 32;
  const int tid = threadIdx.x, wave = tid >> 5, lane = tid & 31;

  if (tid < 128) ax[tid] = ain[(size_t)r0 * 4 + tid];
  __syncthreads();
  for (int idx = tid; idx < 32 * 256; idx += 256) {
    int j = idx >> 8, e = idx & 255;
    float s = b1[e];
    #pragma unroll
    for (int c = 0; c < 4; ++c) s += ax[j * 4 + c] * w1[e * 4 + c];
    u[idx] = s;
  }
  __syncthreads();
  // LN + GELU -> z1 (bf16)
  for (int rr = 0; rr < 4; ++rr) {
    int row = wave * 4 + rr;
    float s = 0.f, s2 = 0.f;
    #pragma unroll
    for (int i = 0; i < 8; ++i) { float x = u[row * 256 + lane + i * 32]; s += x; s2 += x * x; }
    s = wave_sum(s); s2 = wave_sum(s2);
    float m = s * (1.f / 256.f), var = s2 * (1.f / 256.f) - m * m;
    float rs = rsqrtf(var + 1e-5f);
    #pragma unroll
    for (int i = 0; i < 8; ++i) {
      int c = lane + i * 32;
      z1[row * 256 + c] = (bf16)gelu_f((u[row * 256 + c] - m) * rs * l1g[c] + l1b[c]);
    }
  }
  __syncthreads();
  // second linear: 32x256x256 WMMA
  for (int it = 0; it < 4; ++it) {
    int t = wave * 4 + it, mt = t & 1, nt = t >> 1;
    v8f acc = {};
    for (int kk = 0; kk < 8; ++kk) {
      v16bf a = load_a_frag(z1 + mt * 16 * 256 + kk * 32, 256);
      v16bf b = load_b_nk(w2 + nt * 16 * 256 + kk * 32, 256);
      acc = wmma_bf16(a, b, acc);
    }
    int n = lane & 15, mb = (lane & 16) ? 8 : 0, gn = nt * 16 + n;
    #pragma unroll
    for (int r = 0; r < 8; ++r) u[(mt * 16 + mb + r) * 256 + gn] = acc[r] + b2[gn];
  }
  __syncthreads();
  // LN + GELU -> global (bf16)
  for (int rr = 0; rr < 4; ++rr) {
    int row = wave * 4 + rr;
    float s = 0.f, s2 = 0.f;
    #pragma unroll
    for (int i = 0; i < 8; ++i) { float x = u[row * 256 + lane + i * 32]; s += x; s2 += x * x; }
    s = wave_sum(s); s2 = wave_sum(s2);
    float m = s * (1.f / 256.f), var = s2 * (1.f / 256.f) - m * m;
    float rs = rsqrtf(var + 1e-5f);
    #pragma unroll
    for (int i = 0; i < 8; ++i) {
      int c = lane + i * 32;
      zout[(size_t)(r0 + row) * 256 + c] =
          (bf16)gelu_f((u[row * 256 + c] - m) * rs * l2g[c] + l2b[c]);
    }
  }
}

// ---------------------------------------------------------------------------
// Kernel 2b: fusion: gelu(ln(concat(zw,zp) @ fu_w^T + fu_b)) -> comb rows.
// ---------------------------------------------------------------------------
__global__ void fusion_kernel(
    const bf16* __restrict__ zw, const bf16* __restrict__ zp,
    const bf16* __restrict__ fw /* (256,512) */, const float* __restrict__ fb,
    const float* __restrict__ lng, const float* __restrict__ lnb,
    bf16* __restrict__ comb)
{
  extern __shared__ char smem[];
  bf16*  zs = (bf16*)smem;                   // 32x512 concat
  float* u  = (float*)(zs + 32 * 512);       // 32x256
  const int r0 = blockIdx.x * 32;
  const int tid = threadIdx.x, wave = tid >> 5, lane = tid & 31;

  for (int c = tid; c < 2048; c += 256) {    // 16B chunks
    int j = c >> 6, seg = c & 63;
    const bf16* src = (seg < 32) ? (zw + (size_t)(r0 + j) * 256 + seg * 8)
                                 : (zp + (size_t)(r0 + j) * 256 + (seg - 32) * 8);
    cp16(zs + j * 512 + seg * 8, src);
  }
  async_wait0();
  __syncthreads();
  for (int it = 0; it < 4; ++it) {
    int t = wave * 4 + it, mt = t & 1, nt = t >> 1;
    v8f acc = {};
    for (int kk = 0; kk < 16; ++kk) {
      v16bf a = load_a_frag(zs + mt * 16 * 512 + kk * 32, 512);
      v16bf b = load_b_nk(fw + nt * 16 * 512 + kk * 32, 512);
      acc = wmma_bf16(a, b, acc);
    }
    int n = lane & 15, mb = (lane & 16) ? 8 : 0, gn = nt * 16 + n;
    #pragma unroll
    for (int r = 0; r < 8; ++r) u[(mt * 16 + mb + r) * 256 + gn] = acc[r] + fb[gn];
  }
  __syncthreads();
  for (int rr = 0; rr < 4; ++rr) {
    int row = wave * 4 + rr;
    float s = 0.f, s2 = 0.f;
    #pragma unroll
    for (int i = 0; i < 8; ++i) { float x = u[row * 256 + lane + i * 32]; s += x; s2 += x * x; }
    s = wave_sum(s); s2 = wave_sum(s2);
    float m = s * (1.f / 256.f), var = s2 * (1.f / 256.f) - m * m;
    float rs = rsqrtf(var + 1e-5f);
    int gr = r0 + row, bb = gr >> 7, tt = gr & 127;
    #pragma unroll
    for (int i = 0; i < 8; ++i) {
      int c = lane + i * 32;
      comb[((size_t)(bb * 256 + 128 + tt)) * 256 + c] =
          (bf16)gelu_f((u[row * 256 + c] - m) * rs * lng[c] + lnb[c]);
    }
  }
}

// ---------------------------------------------------------------------------
// Kernel 3: generic bf16 WMMA GEMM: out = act(A(M,K) @ W(N,K)^T + bias)[+res]
// 64x64 block tile, async-staged LDS, 8 waves x two 16x16 tiles.
// ---------------------------------------------------------------------------
template <int ACT_GELU, int ADD_RES>
__global__ void gemm_wmma(const bf16* __restrict__ A, const bf16* __restrict__ W,
                          const float* __restrict__ bias, const bf16* __restrict__ res,
                          bf16* __restrict__ out, int M, int N, int K)
{
  __shared__ bf16 As[64 * 32];
  __shared__ bf16 Ws[64 * 32];
  const int n0 = blockIdx.x * 64, m0 = blockIdx.y * 64;
  const int tid = threadIdx.x, wave = tid >> 5, lane = tid & 31;
  const int mt = wave >> 1, ntb = (wave & 1) * 2;
  const int srow = tid >> 2, sseg = tid & 3;
  v8f c0 = {}, c1 = {};
  for (int k0 = 0; k0 < K; k0 += 32) {
    cp16(As + srow * 32 + sseg * 8, A + (size_t)(m0 + srow) * K + k0 + sseg * 8);
    cp16(Ws + srow * 32 + sseg * 8, W + (size_t)(n0 + srow) * K + k0 + sseg * 8);
    async_wait0();
    __syncthreads();
    v16bf a  = load_a_frag(As + mt * 16 * 32, 32);
    v16bf b0 = load_b_nk(Ws + (ntb * 16) * 32, 32);
    v16bf b1 = load_b_nk(Ws + ((ntb + 1) * 16) * 32, 32);
    c0 = wmma_bf16(a, b0, c0);
    c1 = wmma_bf16(a, b1, c1);
    __syncthreads();
  }
  const int n = lane & 15, mb = (lane & 16) ? 8 : 0;
  #pragma unroll
  for (int t = 0; t < 2; ++t) {
    v8f& cc = t ? c1 : c0;
    int gn = n0 + (ntb + t) * 16 + n;
    #pragma unroll
    for (int r = 0; r < 8; ++r) {
      int gm = m0 + mt * 16 + mb + r;
      float v = cc[r] + bias[gn];
      if (ACT_GELU) v = gelu_f(v);
      if (ADD_RES) v += (float)res[(size_t)gm * N + gn];
      out[(size_t)gm * N + gn] = (bf16)v;
    }
  }
}

// ---------------------------------------------------------------------------
// Kernel 4: attention core (seq 256, d=32). One block per (batch, head).
// ---------------------------------------------------------------------------
__global__ void attn_core_kernel(const bf16* __restrict__ qkv /* 8192x768 */,
                                 bf16* __restrict__ aout /* 8192x256 */)
{
  extern __shared__ char smem[];
  bf16*  ks = (bf16*)smem;           // 256x32
  bf16*  vs = ks + 256 * 32;         // 256x32
  bf16*  qs = vs + 256 * 32;         // 64x32 query chunk
  bf16*  ps = qs + 64 * 32;          // 64x256 probs
  float* ss = (float*)(ps + 64 * 256); // 64x256 scores
  const int bh = blockIdx.x, b = bh >> 3, h = bh & 7;
  const int tid = threadIdx.x, wave = tid >> 5, lane = tid & 31;

  {  // stage K and V head slices (async, 16B chunks)
    const bf16* kr = qkv + (size_t)(b * 256 + tid) * 768 + 256 + h * 32;
    const bf16* vr = qkv + (size_t)(b * 256 + tid) * 768 + 512 + h * 32;
    #pragma unroll
    for (int i = 0; i < 4; ++i) {
      cp16(ks + tid * 32 + i * 8, kr + i * 8);
      cp16(vs + tid * 32 + i * 8, vr + i * 8);
    }
  }
  async_wait0();
  __syncthreads();

  const float scale = 0.1767766952966369f;  // 1/sqrt(32)
  for (int qc = 0; qc < 4; ++qc) {
    {  // stage 64-row query chunk
      int row = tid >> 2, seg = tid & 3;
      cp16(qs + row * 32 + seg * 8,
           qkv + (size_t)(b * 256 + qc * 64 + row) * 768 + h * 32 + seg * 8);
    }
    async_wait0();
    __syncthreads();
    // S = Q @ K^T : 4x16 tiles, K=32
    #pragma unroll
    for (int i = 0; i < 8; ++i) {
      int t = wave * 8 + i, mt = t >> 4, nt = t & 15;
      v16bf a = load_a_frag(qs + mt * 16 * 32, 32);
      v16bf bfr = load_b_nk(ks + nt * 16 * 32, 32);
      v8f acc = {};
      acc = wmma_bf16(a, bfr, acc);
      int n = lane & 15, mb = (lane & 16) ? 8 : 0;
      #pragma unroll
      for (int r = 0; r < 8; ++r)
        ss[(mt * 16 + mb + r) * 256 + nt * 16 + n] = acc[r];
    }
    __syncthreads();
    // softmax rows (wave per row group)
    for (int rr = 0; rr < 8; ++rr) {
      int row = wave * 8 + rr;
      float mx = -1e30f;
      #pragma unroll
      for (int i = 0; i < 8; ++i) mx = fmaxf(mx, ss[row * 256 + lane + i * 32]);
      mx = wave_max(mx);
      float e[8], sum = 0.f;
      #pragma unroll
      for (int i = 0; i < 8; ++i) {
        e[i] = __expf((ss[row * 256 + lane + i * 32] - mx) * scale);
        sum += e[i];
      }
      sum = wave_sum(sum);
      float inv = 1.f / sum;
      #pragma unroll
      for (int i = 0; i < 8; ++i)
        ps[row * 256 + lane + i * 32] = (bf16)(e[i] * inv);
    }
    __syncthreads();
    // O = P @ V : 4x2 tiles, K=256
    {
      int mt = wave >> 1, nt = wave & 1;
      v8f acc = {};
      for (int kk = 0; kk < 8; ++kk) {
        v16bf a = load_a_frag(ps + mt * 16 * 256 + kk * 32, 256);
        v16bf bfr = load_b_kn(vs + kk * 32 * 32 + nt * 16, 32);
        acc = wmma_bf16(a, bfr, acc);
      }
      int n = lane & 15, mb = (lane & 16) ? 8 : 0;
      #pragma unroll
      for (int r = 0; r < 8; ++r) {
        int grow = b * 256 + qc * 64 + mt * 16 + mb + r;
        aout[(size_t)grow * 256 + h * 32 + nt * 16 + n] = (bf16)acc[r];
      }
    }
    __syncthreads();
  }
}

// ---------------------------------------------------------------------------
// Kernel 5: row LayerNorm (256-wide rows), one wave per row.
// ---------------------------------------------------------------------------
__global__ void ln_kernel(const bf16* __restrict__ x, const float* __restrict__ g,
                          const float* __restrict__ bv, bf16* __restrict__ y, int rows)
{
  const int row = blockIdx.x * 8 + (threadIdx.x >> 5);
  const int lane = threadIdx.x & 31;
  if (row >= rows) return;
  const bf16* xr = x + (size_t)row * 256;
  float vbuf[8], s = 0.f, s2 = 0.f;
  #pragma unroll
  for (int i = 0; i < 8; ++i) {
    float v = (float)xr[lane + i * 32];
    vbuf[i] = v; s += v; s2 += v * v;
  }
  s = wave_sum(s); s2 = wave_sum(s2);
  float m = s * (1.f / 256.f), var = s2 * (1.f / 256.f) - m * m;
  float rs = rsqrtf(var + 1e-5f);
  #pragma unroll
  for (int i = 0; i < 8; ++i) {
    int c = lane + i * 32;
    y[(size_t)row * 256 + c] = (bf16)((vbuf[i] - m) * rs * g[c] + bv[c]);
  }
}

// ---------------------------------------------------------------------------
// Kernel 6: mean pool + classifier LN + MLP head. One block per batch.
// ---------------------------------------------------------------------------
__global__ void head_kernel(const bf16* __restrict__ t /* 8192x256 */,
                            const float* __restrict__ clg, const float* __restrict__ clb,
                            const float* __restrict__ w1, const float* __restrict__ b1,
                            const float* __restrict__ w2, const float* __restrict__ b2,
                            float* __restrict__ out)
{
  extern __shared__ char smem[];
  float* pooled = (float*)smem;      // 256
  float* zln = pooled + 256;         // 256
  float* h1 = zln + 256;             // 128
  float* red = h1 + 128;             // 16
  const int b = blockIdx.x, tid = threadIdx.x;
  float s = 0.f;
  for (int srow = 0; srow < 256; ++srow)
    s += (float)t[((size_t)(b * 256 + srow)) * 256 + tid];
  pooled[tid] = s * (1.f / 256.f);
  __syncthreads();
  float v = pooled[tid];
  float ws = wave_sum(v), ws2 = wave_sum(v * v);
  int wave = tid >> 5, lane = tid & 31;
  if (lane == 0) { red[wave] = ws; red[wave + 8] = ws2; }
  __syncthreads();
  float m = 0.f, s2 = 0.f;
  #pragma unroll
  for (int i = 0; i < 8; ++i) { m += red[i]; s2 += red[i + 8]; }
  m *= (1.f / 256.f); s2 = s2 * (1.f / 256.f) - m * m;
  float rs = rsqrtf(s2 + 1e-5f);
  zln[tid] = (pooled[tid] - m) * rs * clg[tid] + clb[tid];
  __syncthreads();
  if (tid < 128) {
    float a = b1[tid];
    for (int e = 0; e < 256; ++e) a += zln[e] * w1[tid * 256 + e];
    h1[tid] = gelu_f(a);
  }
  __syncthreads();
  if (tid < 2) {
    float a = b2[tid];
    for (int f = 0; f < 128; ++f) a += h1[f] * w2[tid * 128 + f];
    out[b * 2 + tid] = a;
  }
}

// ---------------------------------------------------------------------------
// Host orchestration
// ---------------------------------------------------------------------------
extern "C" void kernel_launch(void* const* d_in, const int* in_sizes, int n_in,
                              void* d_out, int out_size, void* d_ws, size_t ws_size,
                              hipStream_t stream) {
  (void)in_sizes; (void)n_in; (void)out_size; (void)ws_size;
  int ii = 0;
  const float* in_watch = (const float*)d_in[ii++];
  const float* in_phone = (const float*)d_in[ii++];
  const float* in_skel  = (const float*)d_in[ii++];
  auto F = [&]() { return (const float*)d_in[ii++]; };
  const float *c1_w = F(), *c1_b = F(), *c2_w = F(), *c2_b = F();
  const float *bn1_g = F(), *bn1_b = F(), *bn1_m = F(), *bn1_v = F();
  const float *bn2_g = F(), *bn2_b = F(), *bn2_m = F(), *bn2_v = F();
  const float *ja_wi = F(), *ja_bi = F(), *ja_wo = F(), *ja_bo = F();
  struct AccP { const float *w1, *b1, *l1g, *l1b, *w2, *b2, *l2g, *l2b; } accp[2];
  for (int m = 0; m < 2; ++m) accp[m] = {F(), F(), F(), F(), F(), F(), F(), F()};
  const float *fu_w = F(), *fu_b = F(), *fu_lng = F(), *fu_lnb = F();
  const float *ca_wi = F(), *ca_bi = F(), *ca_wo = F(), *ca_bo = F();
  struct LayerP { const float *wi, *bi, *wo, *bo, *l1g, *l1b, *l2g, *l2b, *w1, *b1, *w2, *b2; } lp[8];
  for (int l = 0; l < 8; ++l)
    lp[l] = {F(), F(), F(), F(), F(), F(), F(), F(), F(), F(), F(), F()};
  const float *enc_g = F(), *enc_b = F();
  const float *cl_g = F(), *cl_b = F(), *cl_w1 = F(), *cl_b1 = F(), *cl_w2 = F(), *cl_b2 = F();

  // bump allocator in workspace (256B aligned)
  char* wsBase = (char*)d_ws;
  size_t off = 0;
  auto abf = [&](size_t n) -> bf16* {
    bf16* p = (bf16*)(wsBase + off);
    off += n * 2; off = (off + 255) & ~(size_t)255;
    return p;
  };
  bf16* w2c_bf  = abf(65536);
  bf16* jawi_bf = abf(196608);
  bf16* jawo_bf = abf(65536);
  bf16* ww2_bf  = abf(65536);
  bf16* pw2_bf  = abf(65536);
  bf16* fuw_bf  = abf(131072);
  bf16* cawi_bf = abf(196608);
  bf16* cawo_bf = abf(65536);
  bf16 *lwi_bf[8], *lwo_bf[8], *lw1_bf[8], *lw2_bf[8];
  for (int l = 0; l < 8; ++l) {
    lwi_bf[l] = abf(196608); lwo_bf[l] = abf(65536);
    lw1_bf[l] = abf(262144); lw2_bf[l] = abf(262144);
  }
  bf16* zw   = abf(1048576);
  bf16* zp   = abf(1048576);
  bf16* xb   = abf(2097152);   // comb / running activation (32,256,256)
  bf16* tb   = abf(2097152);   // LN temp
  bf16* qkvb = abf(6291456);   // (8192,768)
  bf16* ab   = abf(2097152);   // attention output
  bf16* mbuf = abf(8388608);   // MLP hidden (8192,1024)

  // ---- parameter conversion f32 -> bf16 ----
  auto cvt = [&](const float* s, bf16* d, int n) {
    cvt_bf16_kernel<<<(n + 255) / 256, 256, 0, stream>>>(s, d, n);
  };
  cvt_center_kernel<<<256, 256, 0, stream>>>(c2_w, w2c_bf);
  cvt(ja_wi, jawi_bf, 196608); cvt(ja_wo, jawo_bf, 65536);
  cvt(accp[0].w2, ww2_bf, 65536); cvt(accp[1].w2, pw2_bf, 65536);
  cvt(fu_w, fuw_bf, 131072);
  cvt(ca_wi, cawi_bf, 196608); cvt(ca_wo, cawo_bf, 65536);
  for (int l = 0; l < 8; ++l) {
    cvt(lp[l].wi, lwi_bf[l], 196608); cvt(lp[l].wo, lwo_bf[l], 65536);
    cvt(lp[l].w1, lw1_bf[l], 262144); cvt(lp[l].w2, lw2_bf[l], 262144);
  }

  constexpr int SKEL_LDS = 32 * 256 * 2 * 6 + 4 * 32 * 32 * 2 + 4 * 32 * 32 * 4 + 256 * 4 + 96 * 4;
  constexpr int ACC_LDS  = 32 * 256 * 4 + 32 * 256 * 2 + 128 * 4;
  constexpr int FUS_LDS  = 32 * 512 * 2 + 32 * 256 * 4;
  constexpr int ATTN_LDS = 256 * 32 * 2 * 2 + 64 * 32 * 2 + 64 * 256 * 2 + 64 * 256 * 4;
  constexpr int HEAD_LDS = (256 + 256 + 128 + 16) * 4;

  // ---- skeleton branch (fills comb rows [b, 0:128]) ----
  skel_fused_kernel<<<4096, 256, SKEL_LDS, stream>>>(
      in_skel, c1_w, c1_b, bn1_g, bn1_b, bn1_m, bn1_v,
      w2c_bf, c2_b, bn2_g, bn2_b, bn2_m, bn2_v,
      jawi_bf, ja_bi, jawo_bf, ja_bo, xb);

  // ---- accelerometer branches + fusion (fills comb rows [b, 128:256]) ----
  acc_branch_kernel<<<128, 256, ACC_LDS, stream>>>(
      in_watch, accp[0].w1, accp[0].b1, accp[0].l1g, accp[0].l1b,
      ww2_bf, accp[0].b2, accp[0].l2g, accp[0].l2b, zw);
  acc_branch_kernel<<<128, 256, ACC_LDS, stream>>>(
      in_phone, accp[1].w1, accp[1].b1, accp[1].l1g, accp[1].l1b,
      pw2_bf, accp[1].b2, accp[1].l2g, accp[1].l2b, zp);
  fusion_kernel<<<128, 256, FUS_LDS, stream>>>(zw, zp, fuw_bf, fu_b, fu_lng, fu_lnb, xb);

  // ---- cross attention (replaces x, no residual) ----
  gemm_wmma<0, 0><<<dim3(12, 128), 256, 0, stream>>>(xb, cawi_bf, ca_bi, nullptr, qkvb, 8192, 768, 256);
  attn_core_kernel<<<256, 256, ATTN_LDS, stream>>>(qkvb, ab);
  gemm_wmma<0, 0><<<dim3(4, 128), 256, 0, stream>>>(ab, cawo_bf, ca_bo, nullptr, xb, 8192, 256, 256);

  // ---- 8 pre-norm transformer encoder layers ----
  for (int l = 0; l < 8; ++l) {
    ln_kernel<<<1024, 256, 0, stream>>>(xb, lp[l].l1g, lp[l].l1b, tb, 8192);
    gemm_wmma<0, 0><<<dim3(12, 128), 256, 0, stream>>>(tb, lwi_bf[l], lp[l].bi, nullptr, qkvb, 8192, 768, 256);
    attn_core_kernel<<<256, 256, ATTN_LDS, stream>>>(qkvb, ab);
    gemm_wmma<0, 1><<<dim3(4, 128), 256, 0, stream>>>(ab, lwo_bf[l], lp[l].bo, xb, xb, 8192, 256, 256);
    ln_kernel<<<1024, 256, 0, stream>>>(xb, lp[l].l2g, lp[l].l2b, tb, 8192);
    gemm_wmma<1, 0><<<dim3(16, 128), 256, 0, stream>>>(tb, lw1_bf[l], lp[l].b1, nullptr, mbuf, 8192, 1024, 256);
    gemm_wmma<0, 1><<<dim3(4, 128), 256, 0, stream>>>(mbuf, lw2_bf[l], lp[l].b2, xb, xb, 8192, 256, 1024);
  }

  // ---- final norm, pool, classifier ----
  ln_kernel<<<1024, 256, 0, stream>>>(xb, enc_g, enc_b, tb, 8192);
  head_kernel<<<32, 256, HEAD_LDS, stream>>>(tb, cl_g, cl_b, cl_w1, cl_b1, cl_w2, cl_b2, (float*)d_out);
}
